// Attention_63788854280725
// MI455X (gfx1250) — matrix operand order repaired and checked by reading, not compiled, not run on hardware
//
#include <hip/hip_runtime.h>
#include <hip/hip_bf16.h>

#define B_ 128
#define N_ 196
#define E_ 2048
#define D_ 512
#define A_ 512

typedef __attribute__((ext_vector_type(16))) __bf16       v16bf;
typedef __attribute__((ext_vector_type(8)))  float        v8f;
typedef __attribute__((ext_vector_type(4)))  unsigned int u32x4;

union Frag {
    v16bf v;
    u32x4 u[2];
};

__device__ __forceinline__ unsigned short f2bf(float f) {
    unsigned int u = __float_as_uint(f);
    u += 0x7FFFu + ((u >> 16) & 1u);   // round-to-nearest-even
    return (unsigned short)(u >> 16);
}

// Native CDNA5 V_TANH_F32 when the builtin exists; OCML fallback otherwise.
__device__ __forceinline__ float fast_tanh(float x) {
#if __has_builtin(__builtin_amdgcn_tanhf)
    return __builtin_amdgcn_tanhf(x);
#elif __has_builtin(__builtin_amdgcn_tanh_f32)
    return __builtin_amdgcn_tanh_f32(x);
#else
    return tanhf(x);
#endif
}

// Non-temporal 16B load of 4 floats (streaming data bigger than L2).
__device__ __forceinline__ float4 ldg_nt_f4(const float* p) {
    u32x4 u = __builtin_nontemporal_load((const u32x4*)p);
    float4 r;
    r.x = __uint_as_float(u.x);
    r.y = __uint_as_float(u.y);
    r.z = __uint_as_float(u.z);
    r.w = __uint_as_float(u.w);
    return r;
}

// ---------------------------------------------------------------------------
// Kernel 1: repack We [E,A] fp32 -> bf16 in the per-lane B-fragment order used here.
// Fragment layout used here, contiguous k (16x16x32 bf16 B = KxN):
//   lane = col + 16*(kk>>4), element j = kk&15   (kk = k % 32)
// Stored as WeB[((tile*64 + kstep)*32 + lane)*16 + j], tile = a/16, kstep = k/32.
// ---------------------------------------------------------------------------
__global__ __launch_bounds__(256) void conv_we_kernel(const float* __restrict__ We,
                                                      unsigned short* __restrict__ WeB) {
    int idx = blockIdx.x * 256 + threadIdx.x;
    if (idx >= E_ * A_) return;
    int k = idx >> 9;          // / A_
    int a = idx & (A_ - 1);
    unsigned short v = f2bf(We[idx]);
    int tile  = a >> 4;
    int col   = a & 15;
    int kstep = k >> 5;
    int kk    = k & 31;
    int lane  = col + ((kk >> 4) << 4);
    int j     = kk & 15;
    size_t dst = (((size_t)(tile * 64 + kstep) * 32 + lane) << 4) + j;
    WeB[dst] = v;
}

// ---------------------------------------------------------------------------
// Kernel 2: att2[b,a] = decoder_hidden[b,:] @ Wd[:,a] + bd[a]   (tiny GEMV)
// ---------------------------------------------------------------------------
__global__ __launch_bounds__(512) void att2_kernel(const float* __restrict__ h,
                                                   const float* __restrict__ Wd,
                                                   const float* __restrict__ bd,
                                                   float* __restrict__ att2) {
    int b = blockIdx.x;
    int a = threadIdx.x;
    const float* hb = h + b * D_;
    float s = bd[a];
    for (int d = 0; d < D_; ++d)
        s += hb[d] * Wd[d * A_ + a];
    att2[b * A_ + a] = s;
}

// ---------------------------------------------------------------------------
// Kernel 3: fused scores.  One block = (batch b, 16 rows of N).
// 8 waves; wave w owns A-columns [64w, 64w+64).  K loop over E in steps of 32
// using v_wmma_f32_16x16x32_bf16.  Epilogue fuses +att2+be, tanh (native
// v_tanh_f32), dot(Wf), fixed-order LDS reduction -> scores[b,n] (+bf).
// att1 is never materialized.
// ---------------------------------------------------------------------------
#define ROWPAD 4112   // 2048*2 bytes + 16 pad -> conflict-free ds_load_b128

__global__ __launch_bounds__(256) void attn_scores_kernel(
    const float* __restrict__ enc, const unsigned short* __restrict__ WeB,
    const float* __restrict__ att2, const float* __restrict__ be,
    const float* __restrict__ Wf, const float* __restrict__ bfp,
    float* __restrict__ scores) {

    __shared__ __align__(16) unsigned char ldsA[16 * ROWPAD];
    __shared__ float parts[8][32][8];

    const int b   = blockIdx.y;
    const int n0  = blockIdx.x * 16;
    const int tid = threadIdx.x;
    const float* encB = enc + (size_t)b * N_ * E_;

    // ---- stage 16 x 2048 enc tile into LDS as bf16 (rows clamped at N-1) ----
    for (int i = tid; i < 16 * 512; i += 256) {          // 512 float4 per row
        int m  = i >> 9;
        int kq = (i & 511) << 2;
        int n  = n0 + m; if (n > N_ - 1) n = N_ - 1;
        const float4 v = ldg_nt_f4(encB + (size_t)n * E_ + kq);
        unsigned int lo = (unsigned int)f2bf(v.x) | ((unsigned int)f2bf(v.y) << 16);
        unsigned int hi = (unsigned int)f2bf(v.z) | ((unsigned int)f2bf(v.w) << 16);
        unsigned int* dst = (unsigned int*)(ldsA + m * ROWPAD + kq * 2);
        dst[0] = lo;
        dst[1] = hi;
    }
    __syncthreads();

    const int wave = tid >> 5;
    const int lane = tid & 31;
    const int m16  = lane & 15;
    const int lh   = lane >> 4;

    // A fragment (16-bit A 16x32, contiguous k as in WeB): lane<16 holds K 0..15, lane>=16 holds
    // K 16..31 for row M = lane&15.
    const unsigned char* arow = ldsA + m16 * ROWPAD + lh * 32;

    v8f acc[4];
#pragma unroll
    for (int t = 0; t < 4; ++t) acc[t] = (v8f){0.f, 0.f, 0.f, 0.f, 0.f, 0.f, 0.f, 0.f};

    for (int ks = 0; ks < 64; ++ks) {
        Frag a;
        const unsigned char* ap = arow + ks * 64;        // (ks*32)*2 bytes
        a.u[0] = *(const u32x4*)(ap);
        a.u[1] = *(const u32x4*)(ap + 16);

#pragma unroll
        for (int tt = 0; tt < 4; ++tt) {
            const int tile = wave * 4 + tt;
            Frag bf;
            const unsigned short* bp =
                WeB + (((size_t)(tile * 64 + ks) * 32 + lane) << 4);
            bf.u[0] = *(const u32x4*)(bp);
            bf.u[1] = *(const u32x4*)(bp + 8);
            acc[tt] = __builtin_amdgcn_wmma_f32_16x16x32_bf16(
                false, a.v, false, bf.v, (short)0, acc[tt], false, false);
        }
    }

    // ---- epilogue: per-lane fused tanh + dot(Wf) partials --------------------
    // D layout: lane l, VGPR r -> (M = r + 8*(l>>4), N-col = l&15)
    float part[8];
#pragma unroll
    for (int r = 0; r < 8; ++r) part[r] = 0.f;

#pragma unroll
    for (int tt = 0; tt < 4; ++tt) {
        const int a_col = wave * 64 + tt * 16 + m16;
        const float bias = att2[b * A_ + a_col] + be[a_col];
        const float wf   = Wf[a_col];
#pragma unroll
        for (int r = 0; r < 8; ++r)
            part[r] += wf * fast_tanh(acc[tt][r] + bias);
    }
#pragma unroll
    for (int r = 0; r < 8; ++r) parts[wave][lane][r] = part[r];
    __syncthreads();

    // fixed-order reduction -> deterministic
    if (tid < 16) {
        const int m  = tid;
        const int l0 = (m < 8) ? 0 : 16;
        const int r  = m & 7;
        float s = 0.f;
        for (int w = 0; w < 8; ++w)
            for (int l = l0; l < l0 + 16; ++l)
                s += parts[w][l][r];
        const int n = n0 + m;
        if (n < N_) scores[b * 256 + n] = s + bfp[0];
    }
}

// ---------------------------------------------------------------------------
// Kernel 4: per-batch softmax over N + context = sum_n alpha[n] * enc[b,n,:]
// out layout: context [B,E] first, then alpha [B,N].
// enc loads are non-temporal (stream > L2 size).
// ---------------------------------------------------------------------------
__global__ __launch_bounds__(256) void softmax_ctx_kernel(
    const float* __restrict__ enc, const float* __restrict__ scores,
    float* __restrict__ out) {

    __shared__ float sv[256];
    __shared__ float salpha[N_];

    const int b   = blockIdx.x;
    const int tid = threadIdx.x;

    const float s = (tid < N_) ? scores[b * 256 + tid] : -3.4e38f;
    sv[tid] = s;
    __syncthreads();
    for (int off = 128; off > 0; off >>= 1) {
        if (tid < off) sv[tid] = fmaxf(sv[tid], sv[tid + off]);
        __syncthreads();
    }
    const float mx = sv[0];
    __syncthreads();

    const float e = (tid < N_) ? __expf(s - mx) : 0.f;
    sv[tid] = e;
    __syncthreads();
    for (int off = 128; off > 0; off >>= 1) {
        if (tid < off) sv[tid] += sv[tid + off];
        __syncthreads();
    }
    const float inv = 1.f / sv[0];
    if (tid < N_) {
        const float al = e * inv;
        salpha[tid] = al;
        out[(size_t)B_ * E_ + (size_t)b * N_ + tid] = al;   // alpha output
    }
    __syncthreads();

    // context: thread owns 8 consecutive E columns
    const float* eb = enc + (size_t)b * N_ * E_;
    const int e0 = tid * 8;
    float acc0 = 0.f, acc1 = 0.f, acc2 = 0.f, acc3 = 0.f;
    float acc4 = 0.f, acc5 = 0.f, acc6 = 0.f, acc7 = 0.f;
    for (int n = 0; n < N_; ++n) {
        const float al = salpha[n];
        const float* p = eb + (size_t)n * E_ + e0;
        const float4 v0 = ldg_nt_f4(p);
        const float4 v1 = ldg_nt_f4(p + 4);
        acc0 += al * v0.x; acc1 += al * v0.y; acc2 += al * v0.z; acc3 += al * v0.w;
        acc4 += al * v1.x; acc5 += al * v1.y; acc6 += al * v1.z; acc7 += al * v1.w;
    }
    float4* op = (float4*)(out + (size_t)b * E_ + e0);
    op[0] = make_float4(acc0, acc1, acc2, acc3);
    op[1] = make_float4(acc4, acc5, acc6, acc7);
}

// ---------------------------------------------------------------------------
extern "C" void kernel_launch(void* const* d_in, const int* in_sizes, int n_in,
                              void* d_out, int out_size, void* d_ws, size_t ws_size,
                              hipStream_t stream) {
    const float* enc = (const float*)d_in[0];
    const float* h   = (const float*)d_in[1];
    const float* We  = (const float*)d_in[2];
    const float* be  = (const float*)d_in[3];
    const float* Wd  = (const float*)d_in[4];
    const float* bd  = (const float*)d_in[5];
    const float* Wf  = (const float*)d_in[6];
    const float* bfp = (const float*)d_in[7];
    float* out = (float*)d_out;

    // workspace: WeB bf16 (2 MB) | att2 (256 KB) | scores (128 KB)
    unsigned short* WeB   = (unsigned short*)d_ws;
    float*          att2  = (float*)((char*)d_ws + ((size_t)2 << 20));
    float*          score = (float*)((char*)d_ws + ((size_t)2 << 20) + ((size_t)256 << 10));

    hipLaunchKernelGGL(conv_we_kernel, dim3((E_ * A_ + 255) / 256), dim3(256), 0, stream,
                       We, WeB);
    hipLaunchKernelGGL(att2_kernel, dim3(B_), dim3(A_), 0, stream, h, Wd, bd, att2);
    hipLaunchKernelGGL(attn_scores_kernel, dim3((N_ + 15) / 16, B_), dim3(256), 0, stream,
                       enc, WeB, att2, be, Wf, bfp, score);
    hipLaunchKernelGGL(softmax_ctx_kernel, dim3(B_), dim3(256), 0, stream,
                       enc, score, out);
}
